// Net_48498770706963
// MI455X (gfx1250) — compile-verified, matrix operand order checked by reading, not run on hardware
//
#include <hip/hip_runtime.h>
#include <hip/hip_bf16.h>

typedef __attribute__((ext_vector_type(16))) _Float16 v16h;
typedef __attribute__((ext_vector_type(8)))  float    v8f;

#define N_  1024
#define T_  100
#define D_  128
#define S_  20000

#define WMMA_F16(a, b, c) \
    __builtin_amdgcn_wmma_f32_16x16x32_f16(false, (a), false, (b), (short)0, (c), false, false)

// ---------------------------------------------------------------------------
// Kernel 1: embedding-bag sum over T tokens, L2 row-normalize, +bias, ReLU,
// emit f16 query row. One block (128 threads = 4 wave32) per output row.
// ---------------------------------------------------------------------------
__global__ __launch_bounds__(128) void embed_query_kernel(
    const int*   __restrict__ x,          // [N, T]
    const float* __restrict__ emb_table,  // [V_IN+1, D]
    const float* __restrict__ bias,       // [D]
    _Float16*    __restrict__ qh)         // [N, D] f16 out
{
    __shared__ float red[128];
    const int n = blockIdx.x;
    const int d = threadIdx.x;

    const int* xr = x + n * T_;
    float acc = 0.f;
#pragma unroll 4
    for (int t = 0; t < T_; ++t) {
        const int id = xr[t];                       // uniform per block
        acc += emb_table[(size_t)id * D_ + d];      // coalesced 512B row
    }

    red[d] = acc * acc;
    __syncthreads();
#pragma unroll
    for (int off = 64; off > 0; off >>= 1) {
        if (d < off) red[d] += red[d + off];
        __syncthreads();
    }
    const float inv = rsqrtf(red[0]);
    const float q   = fmaxf(acc * inv + bias[d], 0.f);
    qh[(size_t)n * D_ + d] = (_Float16)q;
}

// ---------------------------------------------------------------------------
// Kernel 2: gather sampled class weights to f16 and sampled bias to f32.
// ---------------------------------------------------------------------------
__global__ __launch_bounds__(256) void gather_w_kernel(
    const int*   __restrict__ sample_ids, // [S]
    const float* __restrict__ W,          // [V_OUT, D]
    const float* __restrict__ b_cls,      // [V_OUT]
    _Float16*    __restrict__ wsh,        // [S, D] f16 out
    float*       __restrict__ bs)         // [S] out
{
    const int i   = blockIdx.x * 256 + threadIdx.x;   // 0 .. S*D-1
    const int row = i >> 7;
    const int d   = i & 127;
    const int sid = sample_ids[row];
    wsh[i] = (_Float16)W[(size_t)sid * D_ + d];
    if (d == 0) bs[row] = b_cls[sid];
}

// ---------------------------------------------------------------------------
// Kernel 3: sampled GEMM via v_wmma_f32_16x16x32_f16.
// One wave computes a 64(M) x 32(N) strip: per K-step (32 of 128) it loads
// 4 A fragments + 2 B fragments and issues 8 WMMAs (0.75 loads/WMMA).
// Waves in a block share the same 64-row A quad (hot in WGP$) and sweep
// adjacent N tiles. 10000 waves total, 8 waves (256 thr) per block.
//
// Fragment layouts used here (wave32; contiguous K per lane half, same order for A and B):
//   A 16x32 f16 : lane L holds row M=L&15; K 0-15 (L<16) or 16-31 (L>=16)
//   B 32x16 f16 : lane L holds col N=L&15; K 0-15 (L<16) or 16-31 (L>=16)
//   C/D 16x16 f32: VGPR j, lanes 0-15 -> M=j, N=lane; lanes 16-31 -> M=j+8
// ---------------------------------------------------------------------------
__global__ __launch_bounds__(256) void sampled_gemm_kernel(
    const _Float16* __restrict__ qh,   // [N, D] f16
    const _Float16* __restrict__ wsh,  // [S, D] f16
    const float*    __restrict__ bs,   // [S]
    float*          __restrict__ out)  // [N, S] f32
{
    const int lane  = threadIdx.x & 31;
    const int wave  = blockIdx.x * 8 + (threadIdx.x >> 5);  // 0..9999
    const int mQuad = wave / 625;     // 0..15  : 64 M rows each (N_=1024)
    const int nPair = wave % 625;     // 0..624 : 32 N cols each (S_=20000)

    const int col   = lane & 15;      // N within tile (B/C) or M row (A)
    const int khalf = lane >> 4;      // K half select within fragment
    const int m0    = mQuad * 64;
    const int n0    = nPair * 32;

    const _Float16* bbase = wsh + (size_t)(n0 + col) * D_ + khalf * 16;
    const _Float16* abase = qh  + (size_t)(m0 + col) * D_ + khalf * 16;

    v8f c00 = {}, c01 = {}, c10 = {}, c11 = {};
    v8f c20 = {}, c21 = {}, c30 = {}, c31 = {};
#pragma unroll
    for (int k = 0; k < 4; ++k) {          // K = 128 = 4 * 32
        const v16h b0 = *(const v16h*)(bbase +  0 * D_ + k * 32);
        const v16h b1 = *(const v16h*)(bbase + 16 * D_ + k * 32);
        const v16h a0 = *(const v16h*)(abase +  0 * D_ + k * 32);
        const v16h a1 = *(const v16h*)(abase + 16 * D_ + k * 32);
        const v16h a2 = *(const v16h*)(abase + 32 * D_ + k * 32);
        const v16h a3 = *(const v16h*)(abase + 48 * D_ + k * 32);
        c00 = WMMA_F16(a0, b0, c00);
        c01 = WMMA_F16(a0, b1, c01);
        c10 = WMMA_F16(a1, b0, c10);
        c11 = WMMA_F16(a1, b1, c11);
        c20 = WMMA_F16(a2, b0, c20);
        c21 = WMMA_F16(a2, b1, c21);
        c30 = WMMA_F16(a3, b0, c30);
        c31 = WMMA_F16(a3, b1, c31);
    }

    const float bv0 = bs[n0 + col];
    const float bv1 = bs[n0 + 16 + col];
    // Lane writes M rows {m0 + khalf*8 + j} (+0/16/32/48), cols n0+col (+16).
    float* orow = out + (size_t)(m0 + khalf * 8) * S_ + n0 + col;
#pragma unroll
    for (int j = 0; j < 8; ++j) {
        __builtin_nontemporal_store(c00[j] + bv0, orow + (size_t)(j     ) * S_);
        __builtin_nontemporal_store(c01[j] + bv1, orow + (size_t)(j     ) * S_ + 16);
        __builtin_nontemporal_store(c10[j] + bv0, orow + (size_t)(j + 16) * S_);
        __builtin_nontemporal_store(c11[j] + bv1, orow + (size_t)(j + 16) * S_ + 16);
        __builtin_nontemporal_store(c20[j] + bv0, orow + (size_t)(j + 32) * S_);
        __builtin_nontemporal_store(c21[j] + bv1, orow + (size_t)(j + 32) * S_ + 16);
        __builtin_nontemporal_store(c30[j] + bv0, orow + (size_t)(j + 48) * S_);
        __builtin_nontemporal_store(c31[j] + bv1, orow + (size_t)(j + 48) * S_ + 16);
    }
}

// ---------------------------------------------------------------------------
extern "C" void kernel_launch(void* const* d_in, const int* in_sizes, int n_in,
                              void* d_out, int out_size, void* d_ws, size_t ws_size,
                              hipStream_t stream) {
    const int*   x          = (const int*)  d_in[0];   // [N,T]
    const int*   sample_ids = (const int*)  d_in[1];   // [S]
    const float* emb_table  = (const float*)d_in[2];   // [V_IN+1, D]
    const float* bias       = (const float*)d_in[3];   // [D]
    const float* W          = (const float*)d_in[4];   // [V_OUT, D]
    const float* b_cls      = (const float*)d_in[5];   // [V_OUT]
    float* out = (float*)d_out;                        // [N, S]

    // Workspace layout: qh [N*D] f16 | wsh [S*D] f16 | bs [S] f32  (~5.5 MB)
    _Float16* qh  = (_Float16*)d_ws;
    _Float16* wsh = qh + (size_t)N_ * D_;
    float*    bsx = (float*)(wsh + (size_t)S_ * D_);

    embed_query_kernel<<<N_, 128, 0, stream>>>(x, emb_table, bias, qh);
    gather_w_kernel<<<(S_ * D_) / 256, 256, 0, stream>>>(sample_ids, W, b_cls,
                                                         wsh, bsx);
    // 10000 waves / 8 waves per block = 1250 blocks
    sampled_gemm_kernel<<<1250, 256, 0, stream>>>(qh, wsh, bsx, out);
}
